// Stage_47863115546780
// MI455X (gfx1250) — matrix operand order repaired and checked by reading, not compiled, not run on hardware
//
#include <hip/hip_runtime.h>
#include <math.h>

// =====================================================================
// VRT stage on MI455X (gfx1250, wave32). All dense projections (QKV,
// proj, gated MLPs, lin1/2, fusion MLP) and all 3x3 convs (implicit
// GEMM) run on v_wmma_f32_16x16x32_f16 with f32 accumulation.
// LDS tiles are stored in WMMA *fragment order* so each lane reads its
// 16-half fragment as contiguous 32B (ds_load_b128 x2) instead of 16
// scalar ds_load_u16. Attention (head dim 20) uses a flash-style
// online-softmax kernel with K/V in LDS.
// Workspace requirement: ~190 MB (float arena, see offsets).
// Param indexing assumes insertion-order recursive flatten of the
// setup_inputs() dict: [x, flows_backward, flows_forward, params...].
// =====================================================================

typedef __attribute__((ext_vector_type(16))) _Float16 v16h;
typedef __attribute__((ext_vector_type(2)))  _Float16 v2h;
typedef __attribute__((ext_vector_type(8)))  float    v8f;

#define C_DIM   120
#define N_HEADS 6
#define HDIM    20
#define T_FR    6
#define IMG_H   64
#define IMG_W   64
#define HW_PIX  4096
#define NTOK    24576          // T_FR * HW_PIX
#define MLP_HID 240
#define NPAIR   5              // T-1 frame pairs
#define DEF_KT  1080           // 120 in-ch * 9 taps

#define BM 128                 // rows per block (8 row-tiles, 2 per wave)
#define BN 64                  // cols per block (4 col-tiles)
#define BK 32                  // k-depth per stage (one wmma k-step)

// ------------------------- numeric helpers --------------------------
__device__ inline float geluf(float x){ return 0.5f*x*(1.0f + erff(x*0.70710678118654752f)); }
__device__ inline float lrelu01(float x){ return x > 0.f ? x : 0.1f*x; }
__device__ inline int bandf(int c, int L, int w, int s){ return (c < L - w) ? 0 : ((c < L - s) ? 1 : 2); }

// channel-last bilinear with zero padding (matches reference bilinear_sample)
__device__ inline float bil_cl(const float* __restrict__ img, float py, float px, int c, int C) {
  float x0f = floorf(px), y0f = floorf(py);
  float wx1 = px - x0f, wx0 = 1.f - wx1;
  float wy1 = py - y0f, wy0 = 1.f - wy1;
  int x0 = (int)x0f, y0 = (int)y0f;
  float out = 0.f;
  #pragma unroll
  for (int t = 0; t < 4; ++t) {
    int yi = y0 + (t >> 1), xi = x0 + (t & 1);
    float w = ((t >> 1) ? wy1 : wy0) * ((t & 1) ? wx1 : wx0);
    if (xi >= 0 && xi < IMG_W && yi >= 0 && yi < IMG_H)
      out += w * img[((long)yi*IMG_W + xi)*C + c];
  }
  return out;
}

// fragment-order LDS helpers -----------------------------------------
// A element (r within tile, kk within BK) -> As[r>>4][(r&15)+16*hi][h]
//   hi = (kk>>3)&1 ; h = (kk&7) + 8*(kk>>4). Even-kk pairs share a lane
//   at (h, h+1) -> packed v2h store (ds_store_b32).
// B element (k, c) -> Bs[c>>4][(c&15)+16*((k>>3)&1)][(k&7)+8*(k>>4)]; even-k pairs
//   share a lane -> packed v2h store.

// ------------------------- WMMA GEMM ---------------------------------
// C(M,N) = act(A(M,K) @ B(K,N) + bias) [ep: 0 store, 1 C*=v, 2 C=v+resid]
__global__ __launch_bounds__(128)
void gemm_wmma_kernel(const float* __restrict__ A, const float* __restrict__ Bm,
                      const float* __restrict__ bias, const float* __restrict__ resid,
                      float* __restrict__ C, int M, int N, int K, int act, int ep)
{
  __shared__ _Float16 As[8][32][16];   // 8 row-tiles, fragment order
  __shared__ _Float16 Bs[4][32][16];   // 4 col-tiles, fragment order
  const int tid = threadIdx.x, lane = tid & 31, wave = tid >> 5;
  const int m0 = blockIdx.x * BM, n0 = blockIdx.y * BN;
  const v8f vz = {0.f,0.f,0.f,0.f,0.f,0.f,0.f,0.f};
  v8f acc[2][4];
  #pragma unroll
  for (int rt = 0; rt < 2; ++rt)
    #pragma unroll
    for (int ct = 0; ct < 4; ++ct) acc[rt][ct] = vz;
  const int kTiles = (K + BK - 1) / BK;
  for (int kt = 0; kt < kTiles; ++kt) {
    const int k0 = kt * BK;
    if (kt + 1 < kTiles) // exercise global_prefetch path
      __builtin_prefetch(&A[(long)(m0 + tid)*K + k0 + BK], 0, 0);
    // A tile: 128x32, write in fragment order, packed pairs along k
    for (int e = tid; e < BM*(BK/2); e += 128) {
      int r = e >> 4, p = e & 15;
      int kk = p << 1;
      int gm = m0 + r, gk = k0 + kk;
      float va = 0.f, vb = 0.f;
      if (gm < M) {
        const float* ar = A + (long)gm*K;
        if (gk < K)     va = ar[gk];
        if (gk + 1 < K) vb = ar[gk + 1];
      }
      int hi = (kk >> 3) & 1;
      int h  = (kk & 7) + ((kk >> 4) << 3);
      v2h pr; pr[0] = (_Float16)va; pr[1] = (_Float16)vb;
      *(v2h*)&As[r >> 4][(r & 15) + (hi << 4)][h] = pr;
    }
    // B tile: 32x64, packed pairs along k
    for (int e = tid; e < (BK/2)*BN; e += 128) {
      int c = e & 63, kp = e >> 6;
      int k = kp << 1;
      int gk = k0 + k, gn = n0 + c;
      float va = 0.f, vb = 0.f;
      if (gn < N) {
        if (gk < K)     va = Bm[(long)gk*N + gn];
        if (gk + 1 < K) vb = Bm[(long)(gk + 1)*N + gn];
      }
      v2h pr; pr[0] = (_Float16)va; pr[1] = (_Float16)vb;
      *(v2h*)&Bs[c >> 4][(c & 15) + (((k >> 3) & 1) << 4)][(k & 7) + ((k >> 4) << 3)] = pr;
    }
    __syncthreads();
    v16h bf0 = *(const v16h*)&Bs[0][lane][0];
    v16h bf1 = *(const v16h*)&Bs[1][lane][0];
    v16h bf2 = *(const v16h*)&Bs[2][lane][0];
    v16h bf3 = *(const v16h*)&Bs[3][lane][0];
    #pragma unroll
    for (int rt = 0; rt < 2; ++rt) {
      v16h af = *(const v16h*)&As[wave*2 + rt][lane][0];
      acc[rt][0] = __builtin_amdgcn_wmma_f32_16x16x32_f16(false, af, false, bf0, (short)0, acc[rt][0], false, false);
      acc[rt][1] = __builtin_amdgcn_wmma_f32_16x16x32_f16(false, af, false, bf1, (short)0, acc[rt][1], false, false);
      acc[rt][2] = __builtin_amdgcn_wmma_f32_16x16x32_f16(false, af, false, bf2, (short)0, acc[rt][2], false, false);
      acc[rt][3] = __builtin_amdgcn_wmma_f32_16x16x32_f16(false, af, false, bf3, (short)0, acc[rt][3], false, false);
    }
    __syncthreads();
  }
  #pragma unroll
  for (int rt = 0; rt < 2; ++rt) {
    #pragma unroll
    for (int ct = 0; ct < 4; ++ct) {
      int gn = n0 + ct*16 + (lane & 15);
      #pragma unroll
      for (int r = 0; r < 8; ++r) {
        int gm = m0 + (wave*2 + rt)*16 + r + 8*(lane >> 4);
        if (gm < M && gn < N) {
          float v = acc[rt][ct][r];
          if (bias) v += bias[gn];
          if (act == 1) v = geluf(v); else if (act == 2) v = lrelu01(v);
          long idx = (long)gm*N + gn;
          if (ep == 1) C[idx] *= v;
          else if (ep == 2) C[idx] = v + resid[idx];
          else C[idx] = v;
        }
      }
    }
  }
}

// ---------------- implicit-GEMM 3x3 conv (pad 1), channel-last -------
// In: (img, H, W, Cin)  Bm: (Cin*9, Cout) repacked  Out: (img, H, W, Cout)
__global__ __launch_bounds__(128)
void conv3_wmma_kernel(const float* __restrict__ In, const float* __restrict__ Bm,
                       const float* __restrict__ bias, float* __restrict__ Out,
                       int Cin, int Cout, int act)
{
  __shared__ _Float16 As[8][32][16];
  __shared__ _Float16 Bs[4][32][16];
  const int tid = threadIdx.x, lane = tid & 31, wave = tid >> 5;
  const int m0 = blockIdx.x * BM, n0 = blockIdx.y * BN, img = blockIdx.z;
  const int K = Cin * 9;
  const v8f vz = {0.f,0.f,0.f,0.f,0.f,0.f,0.f,0.f};
  v8f acc[2][4];
  #pragma unroll
  for (int rt = 0; rt < 2; ++rt)
    #pragma unroll
    for (int ct = 0; ct < 4; ++ct) acc[rt][ct] = vz;
  const int kTiles = (K + BK - 1) / BK;
  for (int kt = 0; kt < kTiles; ++kt) {
    const int k0 = kt * BK;
    for (int e = tid; e < BM*(BK/2); e += 128) {
      int r = e >> 4, p = e & 15;
      int kk = p << 1;
      int pix = m0 + r;
      float va = 0.f, vb = 0.f;
      #pragma unroll
      for (int u = 0; u < 2; ++u) {
        int gk = k0 + kk + u;
        if (gk < K) {
          int kpos = gk / Cin, c = gk - kpos*Cin;
          int ky = kpos / 3, kx = kpos - ky*3;
          int py = (pix >> 6) + ky - 1, px = (pix & 63) + kx - 1;
          if (py >= 0 && py < IMG_H && px >= 0 && px < IMG_W) {
            float v = In[((long)img*HW_PIX + py*IMG_W + px)*Cin + c];
            if (u == 0) va = v; else vb = v;
          }
        }
      }
      int hi = (kk >> 3) & 1;
      int h  = (kk & 7) + ((kk >> 4) << 3);
      v2h pr; pr[0] = (_Float16)va; pr[1] = (_Float16)vb;
      *(v2h*)&As[r >> 4][(r & 15) + (hi << 4)][h] = pr;
    }
    for (int e = tid; e < (BK/2)*BN; e += 128) {
      int c = e & 63, kp = e >> 6;
      int k = kp << 1;
      int gk = k0 + k, gn = n0 + c;
      float va = 0.f, vb = 0.f;
      if (gn < Cout) {
        if (gk < K)     va = Bm[(long)gk*Cout + gn];
        if (gk + 1 < K) vb = Bm[(long)(gk + 1)*Cout + gn];
      }
      v2h pr; pr[0] = (_Float16)va; pr[1] = (_Float16)vb;
      *(v2h*)&Bs[c >> 4][(c & 15) + (((k >> 3) & 1) << 4)][(k & 7) + ((k >> 4) << 3)] = pr;
    }
    __syncthreads();
    v16h bf0 = *(const v16h*)&Bs[0][lane][0];
    v16h bf1 = *(const v16h*)&Bs[1][lane][0];
    v16h bf2 = *(const v16h*)&Bs[2][lane][0];
    v16h bf3 = *(const v16h*)&Bs[3][lane][0];
    #pragma unroll
    for (int rt = 0; rt < 2; ++rt) {
      v16h af = *(const v16h*)&As[wave*2 + rt][lane][0];
      acc[rt][0] = __builtin_amdgcn_wmma_f32_16x16x32_f16(false, af, false, bf0, (short)0, acc[rt][0], false, false);
      acc[rt][1] = __builtin_amdgcn_wmma_f32_16x16x32_f16(false, af, false, bf1, (short)0, acc[rt][1], false, false);
      acc[rt][2] = __builtin_amdgcn_wmma_f32_16x16x32_f16(false, af, false, bf2, (short)0, acc[rt][2], false, false);
      acc[rt][3] = __builtin_amdgcn_wmma_f32_16x16x32_f16(false, af, false, bf3, (short)0, acc[rt][3], false, false);
    }
    __syncthreads();
  }
  #pragma unroll
  for (int rt = 0; rt < 2; ++rt) {
    #pragma unroll
    for (int ct = 0; ct < 4; ++ct) {
      int gn = n0 + ct*16 + (lane & 15);
      #pragma unroll
      for (int r = 0; r < 8; ++r) {
        int gm = m0 + (wave*2 + rt)*16 + r + 8*(lane >> 4);
        if (gn < Cout) {
          float v = acc[rt][ct][r] + bias[gn];
          if (act == 2) v = lrelu01(v);
          Out[((long)img*HW_PIX + gm)*Cout + gn] = v;
        }
      }
    }
  }
}

// repack OIHW (Cout,Cin,3,3) -> (k = kpos*Cin + c, Cout)
__global__ void repack_kernel(const float* __restrict__ w, float* __restrict__ wrp,
                              int Cin, int Cout)
{
  long idx = (long)blockIdx.x*blockDim.x + threadIdx.x;
  long total = (long)Cin*9*Cout;
  if (idx >= total) return;
  int o = idx % Cout;
  long k = idx / Cout;
  int kpos = (int)(k / Cin), c = (int)(k - (long)kpos*Cin);
  wrp[idx] = w[((long)o*Cin + c)*9 + kpos];
}

// ---------------- LayerNorms ----------------------------------------
// input (C,T,H,W) planar -> token-major (tok, C), LN over C
__global__ void ln_in_kernel(const float* __restrict__ x, const float* __restrict__ g,
                             const float* __restrict__ b, float* __restrict__ out)
{
  int tok = blockIdx.x*blockDim.x + threadIdx.x;
  if (tok >= NTOK) return;
  float mu = 0.f;
  for (int c = 0; c < C_DIM; ++c) mu += x[(long)c*NTOK + tok];
  mu *= (1.f/C_DIM);
  float var = 0.f;
  for (int c = 0; c < C_DIM; ++c) { float d = x[(long)c*NTOK + tok] - mu; var += d*d; }
  var *= (1.f/C_DIM);
  float rs = rsqrtf(var + 1e-5f);
  for (int c = 0; c < C_DIM; ++c)
    out[(long)tok*C_DIM + c] = (x[(long)c*NTOK + tok] - mu)*rs*g[c] + b[c];
}

// token-major LN over last dim
__global__ void ln_tok_kernel(const float* __restrict__ x, const float* __restrict__ g,
                              const float* __restrict__ b, float* __restrict__ out)
{
  int tok = blockIdx.x*blockDim.x + threadIdx.x;
  if (tok >= NTOK) return;
  const float* r = x + (long)tok*C_DIM;
  float mu = 0.f;
  for (int c = 0; c < C_DIM; ++c) mu += r[c];
  mu *= (1.f/C_DIM);
  float var = 0.f;
  for (int c = 0; c < C_DIM; ++c) { float d = r[c] - mu; var += d*d; }
  var *= (1.f/C_DIM);
  float rs = rsqrtf(var + 1e-5f);
  float* o = out + (long)tok*C_DIM;
  for (int c = 0; c < C_DIM; ++c) o[c] = (r[c] - mu)*rs*g[c] + b[c];
}

// ---------------- window partition / reverse (shift folded in) -------
__global__ void gather_win_kernel(const float* __restrict__ src, float* __restrict__ dst,
                                  int wd, int wh, int ww, int sd, int sh, int sw)
{
  long idx = (long)blockIdx.x*blockDim.x + threadIdx.x;
  if (idx >= (long)NTOK*C_DIM) return;
  int c = (int)(idx % C_DIM);
  long wt = idx / C_DIM;
  int Nw = wd*wh*ww;
  int win = (int)(wt / Nw), n = (int)(wt - (long)win*Nw);
  int nWinH = IMG_H/wh, nWinW = IMG_W/ww;
  int winD = win/(nWinH*nWinW);
  int rem = win - winD*nWinH*nWinW;
  int winH = rem/nWinW, winW = rem - winH*nWinW;
  int id = n/(wh*ww), ih = (n/ww)%wh, iw = n%ww;
  int gd = (winD*wd + id + sd) % T_FR;
  int gh = (winH*wh + ih + sh) % IMG_H;
  int gw = (winW*ww + iw + sw) % IMG_W;
  long tok = ((long)gd*IMG_H + gh)*IMG_W + gw;
  dst[idx] = src[tok*C_DIM + c];
}

__global__ void scatter_win_kernel(const float* __restrict__ src, float* __restrict__ X,
                                   int wd, int wh, int ww, int sd, int sh, int sw)
{
  long idx = (long)blockIdx.x*blockDim.x + threadIdx.x;
  if (idx >= (long)NTOK*C_DIM) return;
  int c = (int)(idx % C_DIM);
  long wt = idx / C_DIM;
  int Nw = wd*wh*ww;
  int win = (int)(wt / Nw), n = (int)(wt - (long)win*Nw);
  int nWinH = IMG_H/wh, nWinW = IMG_W/ww;
  int winD = win/(nWinH*nWinW);
  int rem = win - winD*nWinH*nWinW;
  int winH = rem/nWinW, winW = rem - winH*nWinW;
  int id = n/(wh*ww), ih = (n/ww)%wh, iw = n%ww;
  int gd = (winD*wd + id + sd) % T_FR;
  int gh = (winH*wh + ih + sh) % IMG_H;
  int gw = (winW*ww + iw + sw) % IMG_W;
  long tok = ((long)gd*IMG_H + gh)*IMG_W + gw;
  X[tok*C_DIM + c] += src[idx];
}

// xm = xw + tile(pos_bias,(1,2,1))
__global__ void addpb_kernel(const float* __restrict__ xw, const float* __restrict__ pb,
                             float* __restrict__ xm, int Nwin)
{
  long idx = (long)blockIdx.x*blockDim.x + threadIdx.x;
  if (idx >= (long)NTOK*C_DIM) return;
  int c = (int)(idx % C_DIM);
  int n = (int)((idx / C_DIM) % Nwin);
  xm[idx] = xw[idx] + pb[(n & 63)*C_DIM + c];
}

// sine positional encoding (8x8 window, 120 channels)
__global__ void pb_kernel(float* __restrict__ pb)
{
  int idx = blockIdx.x*blockDim.x + threadIdx.x;
  if (idx >= 64*C_DIM) return;
  int p = idx / C_DIM, c = idx % C_DIM;
  int row = p >> 3, col = p & 7;
  float yv = (float)(row + 1) / (8.0f + 1e-6f) * 6.2831853071795864f;
  float xv = (float)(col + 1) / (8.0f + 1e-6f) * 6.2831853071795864f;
  int cc = (c < 60) ? c : c - 60;
  float base = (c < 60) ? yv : xv;
  float dimt = powf(10000.0f, 2.0f*(float)(cc >> 1)/60.0f);
  float v = base / dimt;
  pb[idx] = (cc & 1) ? cosf(v) : sinf(v);
}

// ---------------- flash-style window attention -----------------------
// qkv rows: (win*NW + tok, 360) = [q(120)|k(120)|v(120)], head-major x 20
__global__ void attn_kernel(const float* __restrict__ qkv, const float* __restrict__ rpb,
                            float* __restrict__ outp,
                            int NW, int nQ, int nK,
                            int qOff, int kOff, int outTokOff,
                            int outStride, int outColOff,
                            int wd, int wh, int ww,
                            int useMask, int msd, int msh, int msw)
{
  extern __shared__ float kv[];            // [K | V] : 2*nK*HDIM floats
  const int win = blockIdx.x, head = blockIdx.y;
  const int tid = threadIdx.x;
  const long wbase = (long)win * NW;
  for (int e = tid; e < nK*HDIM; e += blockDim.x) {
    int j = e / HDIM, dd = e - j*HDIM;
    long row = (wbase + kOff + j) * 360;
    kv[e]            = qkv[row + 120 + head*HDIM + dd];
    kv[nK*HDIM + e]  = qkv[row + 240 + head*HDIM + dd];
  }
  __syncthreads();
  const int i = tid;
  if (i >= nQ) return;
  float q[HDIM];
  {
    long row = (wbase + qOff + i) * 360;
    #pragma unroll
    for (int d = 0; d < HDIM; ++d) q[d] = qkv[row + head*HDIM + d];
  }
  const float scale = 0.22360679774997896f;   // 20^-0.5
  int a = qOff + i;
  int ad = a/(wh*ww), ah = (a/ww)%wh, aw = a%ww;
  int winD = 0, winHc = 0, winWc = 0, lq = 0;
  if (useMask) {
    int nWinH = IMG_H/wh, nWinW = IMG_W/ww;
    winD = win/(nWinH*nWinW);
    int rem = win - winD*nWinH*nWinW;
    winHc = rem/nWinW; winWc = rem - winHc*nWinW;
    int mqd = i/(wh*ww), mqh = (i/ww)%wh, mqw = i%ww;
    lq = bandf(winD*wd + mqd, T_FR, wd, msd)*9
       + bandf(winHc*wh + mqh, IMG_H, wh, msh)*3
       + bandf(winWc*ww + mqw, IMG_W, ww, msw);
  }
  float m = -1e30f, l = 0.f, accv[HDIM];
  #pragma unroll
  for (int d = 0; d < HDIM; ++d) accv[d] = 0.f;
  for (int j = 0; j < nK; ++j) {
    float s = 0.f;
    const float* kj = kv + j*HDIM;
    #pragma unroll
    for (int d = 0; d < HDIM; ++d) s += q[d]*kj[d];
    s *= scale;
    if (rpb) {
      int b2 = kOff + j;
      int bd = b2/(wh*ww), bh = (b2/ww)%wh, bw = b2%ww;
      int idx = (ad-bd+wd-1)*((2*wh-1)*(2*ww-1)) + (ah-bh+wh-1)*(2*ww-1) + (aw-bw+ww-1);
      s += rpb[idx*N_HEADS + head];
    }
    if (useMask) {
      int bd = j/(wh*ww), bh = (j/ww)%wh, bw = j%ww;
      int lk = bandf(winD*wd + bd, T_FR, wd, msd)*9
             + bandf(winHc*wh + bh, IMG_H, wh, msh)*3
             + bandf(winWc*ww + bw, IMG_W, ww, msw);
      if (lk != lq) s -= 100.f;
    }
    float mn = fmaxf(m, s);
    float corr = __expf(m - mn);
    float p = __expf(s - mn);
    l = l*corr + p;
    const float* vj = kv + nK*HDIM + j*HDIM;
    #pragma unroll
    for (int d = 0; d < HDIM; ++d) accv[d] = accv[d]*corr + p*vj[d];
    m = mn;
  }
  long orow = (wbase + outTokOff + i)*outStride + outColOff + head*HDIM;
  float inv = 1.f/l;
  #pragma unroll
  for (int d = 0; d < HDIM; ++d) outp[orow + d] = accv[d]*inv;
}

// ---------------- flow warp / PA-deform pieces -----------------------
__global__ void warp_kernel(const float* __restrict__ X, int srcF0,
                            const float* __restrict__ flows, float* __restrict__ warp)
{
  long idx = (long)blockIdx.x*blockDim.x + threadIdx.x;
  if (idx >= (long)NPAIR*HW_PIX*C_DIM) return;
  int c = (int)(idx % C_DIM);
  long r = idx / C_DIM;
  int img = (int)(r / HW_PIX), pix = (int)(r % HW_PIX);
  float dx = flows[(long)img*2*HW_PIX + pix];
  float dy = flows[(long)img*2*HW_PIX + HW_PIX + pix];
  float px = (float)(pix & 63) + dx;
  float py = (float)(pix >> 6) + dy;
  warp[idx] = bil_cl(X + (long)(srcF0 + img)*HW_PIX*C_DIM, py, px, c, C_DIM);
}

// build conv input [warped(120) | cur(120) | dx | dy] channel-last
__global__ void fcat_kernel(const float* __restrict__ warp, const float* __restrict__ X,
                            int curF0, const float* __restrict__ flows,
                            float* __restrict__ fcat)
{
  long idx = (long)blockIdx.x*blockDim.x + threadIdx.x;
  if (idx >= (long)NPAIR*HW_PIX*242) return;
  int c = (int)(idx % 242);
  long r = idx / 242;
  int img = (int)(r / HW_PIX), pix = (int)(r % HW_PIX);
  float v;
  if (c < 120)       v = warp[r*C_DIM + c];
  else if (c < 240)  v = X[((long)(curF0 + img)*HW_PIX + pix)*C_DIM + (c - 120)];
  else if (c == 240) v = flows[(long)img*2*HW_PIX + pix];
  else               v = flows[(long)img*2*HW_PIX + HW_PIX + pix];
  fcat[idx] = v;
}

// offsets = 10*tanh(o1|o2) + tile(flow[::-1]); mask = sigmoid
__global__ void offmask_kernel(const float* __restrict__ c3o, const float* __restrict__ flows,
                               float* __restrict__ off, float* __restrict__ msk)
{
  long idx = (long)blockIdx.x*blockDim.x + threadIdx.x;
  if (idx >= (long)NPAIR*HW_PIX*216) return;
  int j = (int)(idx % 216);
  long pg = idx / 216;                        // img*HW + pix
  int img = (int)(pg >> 12), pix = (int)(pg & 4095);
  float t = 10.0f * tanhf(c3o[pg*324 + j]);
  // comp 0 = dy (flow ch1), comp 1 = dx (flow ch0)
  float fl = flows[(long)img*2*HW_PIX + (((j & 1) == 0) ? HW_PIX : 0) + pix];
  off[pg*216 + j] = t + fl;
  if (j < 108) {
    float mm = c3o[pg*324 + 216 + j];
    msk[pg*108 + j] = 1.f/(1.f + __expf(-mm));
  }
}

// modulated deformable conv: sample to LDS, then 1080-deep dots
__global__ __launch_bounds__(128)
void deform_kernel(const float* __restrict__ X, int srcF0,
                   const float* __restrict__ off, const float* __restrict__ msk,
                   const float* __restrict__ w, const float* __restrict__ b,
                   float* __restrict__ dst, int dstF0)
{
  __shared__ float vals[2*DEF_KT];
  const int img = blockIdx.z;
  const int p0 = blockIdx.x * 2;
  const int tid = threadIdx.x;
  for (int e = tid; e < 2*DEF_KT; e += 128) {
    int p = e / DEF_KT, t = e - p*DEF_KT;
    int pix = p0 + p;
    int g = t / 90, cc = (t/9) % 10, kk = t % 9;
    int ky = kk/3, kx = kk - ky*3;
    long pg = (long)img*HW_PIX + pix;
    float dy = off[pg*216 + (g*9 + kk)*2];
    float dx = off[pg*216 + (g*9 + kk)*2 + 1];
    float py = (float)(pix >> 6) + (float)(ky - 1) + dy;
    float px = (float)(pix & 63) + (float)(kx - 1) + dx;
    float v = bil_cl(X + (long)(srcF0 + img)*HW_PIX*C_DIM, py, px, g*10 + cc, C_DIM);
    vals[e] = v * msk[pg*108 + g*9 + kk];
  }
  __syncthreads();
  for (int o = tid; o < 2*C_DIM; o += 128) {
    int p = o / C_DIM, oc = o - p*C_DIM;
    const float* wr = w + (long)oc*DEF_KT;  // OIHW -> oc*1080 + cin*9 + tap
    const float* vp = vals + p*DEF_KT;
    float s = b[oc];
    for (int t = 0; t < DEF_KT; ++t) s += vp[t]*wr[t];
    dst[((long)(dstF0 + img)*HW_PIX + p0 + p)*C_DIM + oc] = s;
  }
}

// ---------------- misc elementwise -----------------------------------
__global__ void zero_kernel(float* __restrict__ p, long n)
{
  long i = (long)blockIdx.x*blockDim.x + threadIdx.x;
  if (i < n) p[i] = 0.f;
}

__global__ void zc_kernel(const float* __restrict__ X, const float* __restrict__ XB,
                          const float* __restrict__ XF, float* __restrict__ zc)
{
  long idx = (long)blockIdx.x*blockDim.x + threadIdx.x;
  if (idx >= (long)NTOK*360) return;
  int c = (int)(idx % 360);
  long tok = idx / 360;
  float v;
  if (c < 120)      v = X [tok*C_DIM + c];
  else if (c < 240) v = XB[tok*C_DIM + (c-120)];
  else              v = XF[tok*C_DIM + (c-240)];
  zc[idx] = v;
}

__global__ void outtr_kernel(const float* __restrict__ fin, float* __restrict__ out)
{
  long idx = (long)blockIdx.x*blockDim.x + threadIdx.x;
  if (idx >= (long)NTOK*C_DIM) return;
  int c = (int)(idx / NTOK);
  long tok = idx % NTOK;
  out[idx] = fin[tok*C_DIM + c];
}

// =====================================================================
// host-side orchestration
// =====================================================================
static inline unsigned div_up(long n, int b) { return (unsigned)((n + b - 1)/b); }

static void gemm(hipStream_t st, const float* A, const float* B, const float* bias,
                 const float* resid, float* C, int M, int N, int K, int act, int ep)
{
  dim3 gr(div_up(M, BM), div_up(N, BN));
  gemm_wmma_kernel<<<gr, 128, 0, st>>>(A, B, bias, resid, C, M, N, K, act, ep);
}

// workspace offsets (floats); total ~47.55M floats = ~190 MB
#define A_X0    0L
#define A_X1    2949120L
#define A_LN    5898240L
#define A_XW    8847360L
#define A_XM    11796480L
#define A_QKV   14745600L
#define A_QKVM  23592960L
#define A_ATTN  32440320L
#define A_XKEEP 38338560L
#define A_XB    41287680L
#define A_XF    44236800L
#define A_PB    47185920L
#define A_WRP   47193600L
// conv-stage overlay of the (dead) XW..ATTN arenas
#define CV_WARP 8847360L
#define CV_FCAT 11304960L
#define CV_T0   16261120L
#define CV_T1   18718720L
#define CV_C3O  21176320L
#define CV_OFF  27811840L
#define CV_MSK  32235520L

// param leaf indices (insertion-order recursive flatten)
#define P_RN_G 3
#define P_RN_B 4
#define P_G1   5      // 3 mut blocks x 17 leaves
#define P_G2   56     // 1 block x 15 leaves
#define P_LIN1_W 71
#define P_LIN1_B 72
#define P_LIN2_W 73
#define P_LIN2_B 74
#define P_PA   75     // c0_w,c0_b,c1_w,c1_b,c2_w,c2_b,c3_w,c3_b,dcn_w,dcn_b
#define P_FUSE 85     // fc11_w,fc11_b,fc12_w,fc12_b,fc2_w,fc2_b
// mut-block leaf offsets
enum { B_N1G=0, B_N1B, B_QKVW, B_QKVB, B_RPB, B_PROJW, B_PROJB, B_N2G, B_N2B,
       B_F11W, B_F11B, B_F12W, B_F12B, B_F2W, B_F2B, B_QMW, B_QMB };

static void run_block(hipStream_t st, const float* const* pp, int base, int mut,
                      int wd, int wh, int ww, int shifted,
                      float* X, float* LNb, float* XW, float* XM,
                      float* QKV, float* QKVM, float* ATTN, const float* PB)
{
  const int NW = wd*wh*ww;
  const int numWin = (T_FR/wd)*(IMG_H/wh)*(IMG_W/ww);
  const int sd = shifted ? wd/2 : 0, sh = shifted ? wh/2 : 0, sw = shifted ? ww/2 : 0;
  const int msd = wd/2, msh = wh/2, msw = ww/2;
  const long TC = (long)NTOK*C_DIM;
  const int outStride = mut ? 240 : 120;

  ln_tok_kernel<<<div_up(NTOK,256),256,0,st>>>(X, pp[base+B_N1G], pp[base+B_N1B], LNb);
  gather_win_kernel<<<div_up(TC,256),256,0,st>>>(LNb, XW, wd,wh,ww, sd,sh,sw);
  gemm(st, XW, pp[base+B_QKVW], pp[base+B_QKVB], nullptr, QKV, NTOK, 360, C_DIM, 0, 0);
  if (mut) {
    addpb_kernel<<<div_up(TC,256),256,0,st>>>(XW, PB, XM, NW);
    gemm(st, XM, pp[base+B_QMW], pp[base+B_QMB], nullptr, QKVM, NTOK, 360, C_DIM, 0, 0);
  }
  // self attention (rpb bias, mask if shifted) -> ATTN cols [mut?120:0, +120)
  {
    size_t sh_b = (size_t)2*NW*HDIM*sizeof(float);
    attn_kernel<<<dim3(numWin, N_HEADS), NW, sh_b, st>>>(
        QKV, pp[base+B_RPB], ATTN, NW, NW, NW, 0, 0, 0,
        outStride, mut ? 120 : 0, wd, wh, ww, shifted, msd, msh, msw);
  }
  if (mut) {
    const int n2 = NW/2;
    size_t sh_b = (size_t)2*n2*HDIM*sizeof(float);
    // x1 = attend(q2, k1, v1) -> tokens [0,n2), cols [0,120)
    attn_kernel<<<dim3(numWin, N_HEADS), n2, sh_b, st>>>(
        QKVM, nullptr, ATTN, NW, n2, n2, n2, 0, 0,
        240, 0, wd, wh, ww, shifted, msd, msh, msw);
    // x2 = attend(q1, k2, v2) -> tokens [n2,NW), cols [0,120)
    attn_kernel<<<dim3(numWin, N_HEADS), n2, sh_b, st>>>(
        QKVM, nullptr, ATTN, NW, n2, n2, 0, n2, n2,
        240, 0, wd, wh, ww, shifted, msd, msh, msw);
  }
  // proj -> XM, scatter residual-add into X
  gemm(st, ATTN, pp[base+B_PROJW], pp[base+B_PROJB], nullptr, XM,
       NTOK, C_DIM, mut ? 240 : 120, 0, 0);
  scatter_win_kernel<<<div_up(TC,256),256,0,st>>>(XM, X, wd,wh,ww, sd,sh,sw);
  // gated MLP: H = gelu(z@fc11+b) ; H *= z@fc12+b ; X += H@fc2+b
  ln_tok_kernel<<<div_up(NTOK,256),256,0,st>>>(X, pp[base+B_N2G], pp[base+B_N2B], LNb);
  float* H = ATTN;   // reuse attention arena (24576 x 240)
  gemm(st, LNb, pp[base+B_F11W], pp[base+B_F11B], nullptr, H, NTOK, MLP_HID, C_DIM, 1, 0);
  gemm(st, LNb, pp[base+B_F12W], pp[base+B_F12B], nullptr, H, NTOK, MLP_HID, C_DIM, 0, 1);
  gemm(st, H, pp[base+B_F2W], pp[base+B_F2B], X, X, NTOK, C_DIM, MLP_HID, 0, 2);
}

static void conv3(hipStream_t st, const float* In, const float* W, const float* bias,
                  float* wrp, float* Out, int Cin, int Cout, int act)
{
  long kn = (long)Cin*9*Cout;
  repack_kernel<<<div_up(kn,256),256,0,st>>>(W, wrp, Cin, Cout);
  dim3 gr(HW_PIX/BM, div_up(Cout, BN), NPAIR);
  conv3_wmma_kernel<<<gr, 128, 0, st>>>(In, wrp, bias, Out, Cin, Cout, act);
}

static void run_align_dir(hipStream_t st, const float* const* pp, float* ws,
                          const float* flows, int srcF0, int curF0,
                          float* dst, int dstF0, int zeroFrame, const float* X)
{
  float* WARP = ws + CV_WARP;  float* FCAT = ws + CV_FCAT;
  float* T0   = ws + CV_T0;    float* T1   = ws + CV_T1;
  float* C3O  = ws + CV_C3O;   float* OFF  = ws + CV_OFF;
  float* MSK  = ws + CV_MSK;   float* WRP  = ws + A_WRP;
  long nWarp = (long)NPAIR*HW_PIX*C_DIM;
  warp_kernel<<<div_up(nWarp,256),256,0,st>>>(X, srcF0, flows, WARP);
  long nF = (long)NPAIR*HW_PIX*242;
  fcat_kernel<<<div_up(nF,256),256,0,st>>>(WARP, X, curF0, flows, FCAT);
  conv3(st, FCAT, pp[P_PA+0], pp[P_PA+1], WRP, T0, 242, 120, 2);
  conv3(st, T0,   pp[P_PA+2], pp[P_PA+3], WRP, T1, 120, 120, 2);
  conv3(st, T1,   pp[P_PA+4], pp[P_PA+5], WRP, T0, 120, 120, 2);
  conv3(st, T0,   pp[P_PA+6], pp[P_PA+7], WRP, C3O, 120, 324, 0);
  long nOM = (long)NPAIR*HW_PIX*216;
  offmask_kernel<<<div_up(nOM,256),256,0,st>>>(C3O, flows, OFF, MSK);
  deform_kernel<<<dim3(HW_PIX/2, 1, NPAIR), 128, 0, st>>>(
      X, srcF0, OFF, MSK, pp[P_PA+8], pp[P_PA+9], dst, dstF0);
  zero_kernel<<<div_up((long)HW_PIX*C_DIM,256),256,0,st>>>(
      dst + (long)zeroFrame*HW_PIX*C_DIM, (long)HW_PIX*C_DIM);
}

extern "C" void kernel_launch(void* const* d_in, const int* in_sizes, int n_in,
                              void* d_out, int out_size, void* d_ws, size_t ws_size,
                              hipStream_t stream)
{
  (void)in_sizes; (void)out_size; (void)ws_size;
  const float* pp[128];
  int nn = n_in < 128 ? n_in : 128;
  for (int i = 0; i < nn; ++i) pp[i] = (const float*)d_in[i];

  float* ws = (float*)d_ws;
  float* X0 = ws + A_X0;   float* X1 = ws + A_X1;   float* LNb = ws + A_LN;
  float* XW = ws + A_XW;   float* XM = ws + A_XM;
  float* QKV = ws + A_QKV; float* QKVM = ws + A_QKVM;
  float* ATTN = ws + A_ATTN;
  float* XKEEP = ws + A_XKEEP;
  float* XB = ws + A_XB;   float* XF = ws + A_XF;
  float* PB = ws + A_PB;
  const long TCbytes = (long)NTOK*C_DIM*sizeof(float);

  // positional encoding + input LN
  pb_kernel<<<div_up(64L*C_DIM,256),256,0,stream>>>(PB);
  ln_in_kernel<<<div_up(NTOK,256),256,0,stream>>>(pp[0], pp[P_RN_G], pp[P_RN_B], X0);

  float* X = X0; float* Xalt = X1;

  // ---- group 1: ws=(2,8,8), mutual attention, blocks [noshift, shift, noshift]
  hipMemcpyAsync(XKEEP, X, TCbytes, hipMemcpyDeviceToDevice, stream);
  for (int blk = 0; blk < 3; ++blk)
    run_block(stream, pp, P_G1 + blk*17, 1, 2, 8, 8, blk & 1,
              X, LNb, XW, XM, QKV, QKVM, ATTN, PB);
  gemm(stream, X, pp[P_LIN1_W], pp[P_LIN1_B], XKEEP, Xalt, NTOK, C_DIM, C_DIM, 0, 2);
  { float* t = X; X = Xalt; Xalt = t; }

  // ---- group 2: ws=(6,8,8), single block, no shift, no mut
  hipMemcpyAsync(XKEEP, X, TCbytes, hipMemcpyDeviceToDevice, stream);
  run_block(stream, pp, P_G2, 0, 6, 8, 8, 0,
            X, LNb, XW, XM, QKV, QKVM, ATTN, PB);
  gemm(stream, X, pp[P_LIN2_W], pp[P_LIN2_B], XKEEP, Xalt, NTOK, C_DIM, C_DIM, 0, 2);
  { float* t = X; X = Xalt; Xalt = t; }

  // ---- parallel-warping alignment (flow warp + PA-deform conv)
  // backward: src frames 1..5, ref 0..4 -> XB slots 0..4, slot 5 zero
  run_align_dir(stream, pp, ws, pp[1], 1, 0, XB, 0, T_FR - 1, X);
  // forward: src frames 0..4, ref 1..5 -> XF slots 1..5, slot 0 zero
  run_align_dir(stream, pp, ws, pp[2], 0, 1, XF, 1, 0, X);

  // ---- fusion gated MLP over [x | xb | xf] (360 -> 120)
  float* ZC = ws + A_QKV;      // conv arenas dead now
  float* FH = ws + A_QKVM;
  float* FIN = ws + A_LN;
  long nZ = (long)NTOK*360;
  zc_kernel<<<div_up(nZ,256),256,0,stream>>>(X, XB, XF, ZC);
  gemm(stream, ZC, pp[P_FUSE+0], pp[P_FUSE+1], nullptr, FH, NTOK, 360, 360, 1, 0);
  gemm(stream, ZC, pp[P_FUSE+2], pp[P_FUSE+3], nullptr, FH, NTOK, 360, 360, 0, 1);
  gemm(stream, FH, pp[P_FUSE+4], pp[P_FUSE+5], nullptr, FIN, NTOK, C_DIM, 360, 0, 0);

  // ---- (B,T,H,W,C) -> (B,C,T,H,W)
  outtr_kernel<<<div_up((long)NTOK*C_DIM,256),256,0,stream>>>(FIN, (float*)d_out);
}
